// DeepSeekV32Attention_12627203850987
// MI455X (gfx1250) — matrix operand order repaired and checked by reading, compiled, not run on hardware
//
#include <hip/hip_runtime.h>
#include <hip/hip_bf16.h>

// ---------------------------------------------------------------------------
// DeepSeek MLA attention for MI455X (gfx1250, wave32).
// - All GEMMs on v_wmma_f32_16x16x32_bf16 (fp32 accumulate)
// - Weights pre-transposed to bf16 so GEMM tiles stage via
//   GLOBAL_LOAD_ASYNC_TO_LDS_B128 (ASYNCcnt path), overlapping WMMA
// - Flash-style streaming attention, softmax/rmsnorm in fp32
// ---------------------------------------------------------------------------

typedef unsigned int  u32;
typedef unsigned short u16;
typedef __attribute__((ext_vector_type(16))) __bf16 bf16x16;
typedef __attribute__((ext_vector_type(8)))  float  f32x8;

union Frag {
    bf16x16 v;
    u32 u[8];
};

#if defined(__gfx1250__) && __has_builtin(__builtin_amdgcn_global_load_async_to_lds_b128)
#define USE_ASYNC 1
#else
#define USE_ASYNC 0
#endif

#if USE_ASYNC
typedef int v4i __attribute__((vector_size(16)));
typedef __attribute__((address_space(1))) v4i g_v4i;   // global (prints as __device__)
typedef __attribute__((address_space(3))) v4i l_v4i;   // LDS    (prints as __shared__)
__device__ __forceinline__ void async_b128(const void* g, void* l) {
    __builtin_amdgcn_global_load_async_to_lds_b128(
        (g_v4i*)const_cast<void*>(g), (l_v4i*)l, 0, 0);
}
__device__ __forceinline__ void wait_async0() {
#if __has_builtin(__builtin_amdgcn_s_wait_asynccnt)
    __builtin_amdgcn_s_wait_asynccnt(0);
#else
    asm volatile("s_wait_asynccnt 0x0" ::: "memory");
#endif
}
#else
__device__ __forceinline__ void wait_async0() {}
#endif

__device__ __forceinline__ u16 f2bf(float f) {
    u32 x = __builtin_bit_cast(u32, f);
    x += 0x7fffu + ((x >> 16) & 1u);      // round-to-nearest-even
    return (u16)(x >> 16);
}

__device__ __forceinline__ f32x8 wmma_bf16(const Frag& a, const Frag& b, f32x8 c) {
    return __builtin_amdgcn_wmma_f32_16x16x32_bf16(
        false, a.v, false, b.v, (short)0, c, false, false);
}

// Reductions within a 16-lane half (rows M=r / M=r+8 live in separate halves
// of the C layout; xor masks < 16 never cross halves).
__device__ __forceinline__ float half_max(float v) {
    v = fmaxf(v, __shfl_xor(v, 1, 32));
    v = fmaxf(v, __shfl_xor(v, 2, 32));
    v = fmaxf(v, __shfl_xor(v, 4, 32));
    v = fmaxf(v, __shfl_xor(v, 8, 32));
    return v;
}
__device__ __forceinline__ float half_sum(float v) {
    v += __shfl_xor(v, 1, 32);
    v += __shfl_xor(v, 2, 32);
    v += __shfl_xor(v, 4, 32);
    v += __shfl_xor(v, 8, 32);
    return v;
}

// ---------------------------------------------------------------------------
// fp32 -> bf16 conversion (row-major preserved)
// ---------------------------------------------------------------------------
__global__ __launch_bounds__(256) void cvt_f32_bf16_kernel(
    const float* __restrict__ in, u16* __restrict__ out, int n4) {
    int i = blockIdx.x * 256 + threadIdx.x;
    if (i < n4) {
        float4 v = ((const float4*)in)[i];
        uint2 p;
        p.x = (u32)f2bf(v.x) | ((u32)f2bf(v.y) << 16);
        p.y = (u32)f2bf(v.z) | ((u32)f2bf(v.w) << 16);
        ((uint2*)out)[i] = p;
    }
}

// fp32 W[K][N] -> bf16 WT[N][K]  (tiled 32x32, both sides coalesced)
__global__ __launch_bounds__(256) void cvt_transpose_kernel(
    const float* __restrict__ in, u16* __restrict__ out, int K, int N) {
    __shared__ float tile[32][33];
    const int kb = blockIdx.y * 32, nb = blockIdx.x * 32;
    const int tr = threadIdx.x >> 5, tc = threadIdx.x & 31;
#pragma unroll
    for (int i = 0; i < 4; ++i) {
        const int r = tr + i * 8;
        tile[r][tc] = in[(size_t)(kb + r) * N + nb + tc];
    }
    __syncthreads();
#pragma unroll
    for (int i = 0; i < 4; ++i) {
        const int r = tr + i * 8;
        out[(size_t)(nb + r) * K + kb + tc] = f2bf(tile[tc][r]);
    }
}

// ---------------------------------------------------------------------------
// Row-wise RMSNorm: fp32 in (stride strideIn) -> bf16 out (dense width W)
// ---------------------------------------------------------------------------
__global__ __launch_bounds__(256) void rmsnorm_kernel(
    const float* __restrict__ in, const float* __restrict__ w,
    u16* __restrict__ out, int W, int strideIn) {
    __shared__ float red[256];
    const int row = blockIdx.x;
    const float* x = in + (size_t)row * strideIn;
    float ss = 0.f;
    for (int i = threadIdx.x; i < W; i += 256) { float v = x[i]; ss += v * v; }
    red[threadIdx.x] = ss;
    __syncthreads();
    for (int off = 128; off > 0; off >>= 1) {
        if ((int)threadIdx.x < off) red[threadIdx.x] += red[threadIdx.x + off];
        __syncthreads();
    }
    const float inv = rsqrtf(red[0] / (float)W + 1e-6f);
    for (int i = threadIdx.x; i < W; i += 256)
        out[(size_t)row * W + i] = f2bf(x[i] * inv * w[i]);
}

// Extract k_rope (cols 512..575 of kv_c) -> bf16 [S][64]
__global__ __launch_bounds__(256) void krope_kernel(
    const float* __restrict__ kvc, u16* __restrict__ kr) {
    int i = blockIdx.x * 256 + threadIdx.x;   // S*64 total
    int r = i >> 6, c = i & 63;
    kr[i] = f2bf(kvc[(size_t)r * 576 + 512 + c]);
}

// ---------------------------------------------------------------------------
// bf16 WMMA GEMM: C[M,N] = A[M,K] * B[K,N], with B given TRANSPOSED (BT[N][K]).
// Block tile 128(M) x 64(N), K-step 32, 8 waves, double-buffered LDS staged
// with async global->LDS b128 copies. M%128==0, N%64==0, K%32==0.
// ---------------------------------------------------------------------------
template <bool OUT_BF16>
__global__ __launch_bounds__(256) void gemm_bf16_kernel(
    const u16* __restrict__ A, const u16* __restrict__ BT,
    void* __restrict__ Cout, int M, int N, int K) {
    __shared__ u16 lA[2][128 * 40];  // [m][k], stride 40 (80B rows, 16B aligned)
    __shared__ u16 lB[2][64 * 40];   // [n][k], stride 40

    const int t   = threadIdx.x;
    const int m0  = blockIdx.y * 128;
    const int n0  = blockIdx.x * 64;
    const int wid = t >> 5, lane = t & 31;
    const int wm  = wid >> 1, wn = wid & 1;       // 4x2 wave grid
    const int hl  = lane >> 4, ln = lane & 15;

    auto stage = [&](int buf, int kk) {
#if USE_ASYNC
#pragma unroll
        for (int i = 0; i < 2; ++i) {             // A: 128 rows x 64B
            const int idx = t + i * 256;          // 512 x b128
            const int r = idx >> 2, seg = idx & 3;
            async_b128(A + (size_t)(m0 + r) * K + kk + seg * 8,
                       &lA[buf][r * 40 + seg * 8]);
        }
        {                                         // B: 64 rows x 64B -> 256 x b128
            const int n = t >> 2, seg = t & 3;
            async_b128(BT + (size_t)(n0 + n) * K + kk + seg * 8,
                       &lB[buf][n * 40 + seg * 8]);
        }
#else
        {
            const int r = t >> 1, c0 = (t & 1) << 4;
            const u32* g = (const u32*)(A + (size_t)(m0 + r) * K + kk + c0);
            u32* s = (u32*)&lA[buf][r * 40 + c0];
#pragma unroll
            for (int i = 0; i < 8; ++i) s[i] = g[i];
        }
#pragma unroll
        for (int i = 0; i < 4; ++i) {
            const int idx = t + i * 256;          // 1024 u32
            const int n = idx >> 4, cu = idx & 15;
            *(u32*)&lB[buf][n * 40 + cu * 2] =
                *(const u32*)(BT + (size_t)(n0 + n) * K + kk + cu * 2);
        }
#endif
    };

    f32x8 acc[2][2];
#pragma unroll
    for (int mt = 0; mt < 2; ++mt)
#pragma unroll
        for (int nt = 0; nt < 2; ++nt)
#pragma unroll
            for (int r = 0; r < 8; ++r) acc[mt][nt][r] = 0.f;

    stage(0, 0);
    const int nk = K >> 5;
    for (int s = 0; s < nk; ++s) {
        const int buf = s & 1;
        wait_async0();
        __syncthreads();   // stage(s) visible; all reads of buf^1 finished
        if (s + 1 < nk) stage(buf ^ 1, (s + 1) * 32);   // overlaps WMMA below

        Frag a[2], b[2];
#pragma unroll
        for (int mt = 0; mt < 2; ++mt) {
            const int ml = wm * 32 + mt * 16 + ln;
#pragma unroll
            for (int v = 0; v < 8; ++v) {
                const int K0 = ((v >> 2) << 4) + hl * 8 + ((v & 3) << 1);
                a[mt].u[v] = *(const u32*)&lA[buf][ml * 40 + K0];
            }
        }
#pragma unroll
        for (int nt = 0; nt < 2; ++nt) {
            const int nl = wn * 32 + nt * 16 + ln;
#pragma unroll
            for (int v = 0; v < 8; ++v) {
                const int kk = ((v >> 2) << 4) + hl * 8 + ((v & 3) << 1);
                b[nt].u[v] = *(const u32*)&lB[buf][nl * 40 + kk];
            }
        }
#pragma unroll
        for (int mt = 0; mt < 2; ++mt)
#pragma unroll
            for (int nt = 0; nt < 2; ++nt)
                acc[mt][nt] = wmma_bf16(a[mt], b[nt], acc[mt][nt]);
    }

#pragma unroll
    for (int mt = 0; mt < 2; ++mt)
#pragma unroll
        for (int nt = 0; nt < 2; ++nt)
#pragma unroll
            for (int r = 0; r < 8; ++r) {
                const int row = m0 + wm * 32 + mt * 16 + hl * 8 + r;
                const int col = n0 + wn * 32 + nt * 16 + ln;
                if (OUT_BF16)
                    ((u16*)Cout)[(size_t)row * N + col] = f2bf(acc[mt][nt][r]);
                else
                    ((float*)Cout)[(size_t)row * N + col] = acc[mt][nt][r];
            }
}

// ---------------------------------------------------------------------------
// Flash MLA attention. grid = (S/128 q-blocks, 16 heads), 256 threads (8 waves).
// Each wave owns 16 query rows; streams 64-key blocks (causal).
//  Q : [S][16*192] bf16   KV : [S][16*256] bf16 (128 k_nope | 128 v)
//  KR: [S][64]     bf16   O  : [S][16*128] bf16
// ---------------------------------------------------------------------------
__global__ __launch_bounds__(256) void mla_attn_kernel(
    const u16* __restrict__ Q, const u16* __restrict__ KV,
    const u16* __restrict__ KR, u16* __restrict__ O) {
    __shared__ u16 lK[64 * 200];       // [key][0..191] (nope|rope), stride 200 (400B)
    __shared__ u16 lVT[128 * 68];      // [d][key], stride 68
    __shared__ u16 lP[8 * 16 * 68];    // per-wave P strips [16][key], stride 68

    const int qb = blockIdx.x, h = blockIdx.y;
    const int q0 = qb * 128;
    const int t = threadIdx.x, wid = t >> 5, lane = t & 31;
    const int hl = lane >> 4, ln = lane & 15;

    // Q fragments: 16 rows x 192, in registers (6 k-steps of 32).
    Frag aq[6];
    {
        const u16* qp = Q + (size_t)(q0 + wid * 16 + ln) * 3072 + h * 192;
#pragma unroll
        for (int ks = 0; ks < 6; ++ks)
#pragma unroll
            for (int v = 0; v < 8; ++v) {
                const int K0 = ks * 32 + ((v >> 2) << 4) + hl * 8 + ((v & 3) << 1);
                aq[ks].u[v] = *(const u32*)(qp + K0);
            }
    }

    float m_i[8], l_i[8];
    f32x8 o_acc[8];
#pragma unroll
    for (int r = 0; r < 8; ++r) { m_i[r] = -1e30f; l_i[r] = 0.f; }
#pragma unroll
    for (int a = 0; a < 8; ++a)
#pragma unroll
        for (int r = 0; r < 8; ++r) o_acc[a][r] = 0.f;

    const float scale = 0.07216878364870323f;  // 1/sqrt(192)
    const int kbmax = 2 * qb + 1;

    for (int kb = 0; kb <= kbmax; ++kb) {
        __syncthreads();  // all reads of lK/lVT from previous iteration done
        // ---- stage K head tile (contiguous rows -> async) ----
#if USE_ASYNC
#pragma unroll
        for (int i = 0; i < 4; ++i) {            // nope: 64 rows x 256B
            const int idx = t + i * 256;         // 1024 x b128
            const int kr = idx >> 4, seg = idx & 15;
            async_b128(KV + (size_t)(kb * 64 + kr) * 4096 + h * 256 + seg * 8,
                       &lK[kr * 200 + seg * 8]);
        }
#pragma unroll
        for (int i = 0; i < 2; ++i) {            // rope: 64 rows x 128B
            const int idx = t + i * 256;         // 512 x b128
            const int kr = idx >> 3, seg = idx & 7;
            async_b128(KR + (size_t)(kb * 64 + kr) * 64 + seg * 8,
                       &lK[kr * 200 + 128 + seg * 8]);
        }
#else
#pragma unroll
        for (int i = 0; i < 16; ++i) {
            const int idx = t + i * 256;
            const int kr = idx >> 6, cu = idx & 63;
            *(u32*)&lK[kr * 200 + cu * 2] =
                *(const u32*)(KV + (size_t)(kb * 64 + kr) * 4096 + h * 256 + cu * 2);
        }
#pragma unroll
        for (int i = 0; i < 8; ++i) {
            const int idx = t + i * 256;
            const int kr = idx >> 5, cu = idx & 31;
            *(u32*)&lK[kr * 200 + 128 + cu * 2] =
                *(const u32*)(KR + (size_t)(kb * 64 + kr) * 64 + cu * 2);
        }
#endif
        // ---- stage V transposed: lVT[d][key] (manual transpose) ----
#pragma unroll
        for (int i = 0; i < 16; ++i) {           // 64 keys x 64 d-pairs
            const int idx = t + i * 256;
            const int key = idx >> 6, du = idx & 63;
            const u32 pr =
                *(const u32*)(KV + (size_t)(kb * 64 + key) * 4096 + h * 256 + 128 + du * 2);
            lVT[(du * 2 + 0) * 68 + key] = (u16)pr;
            lVT[(du * 2 + 1) * 68 + key] = (u16)(pr >> 16);
        }
        wait_async0();
        __syncthreads();

        // ---- S = Q @ K^T : strip 16 x 64 per wave ----
        f32x8 s4[4];
#pragma unroll
        for (int nt = 0; nt < 4; ++nt)
#pragma unroll
            for (int r = 0; r < 8; ++r) s4[nt][r] = 0.f;
#pragma unroll
        for (int nt = 0; nt < 4; ++nt)
#pragma unroll
            for (int ks = 0; ks < 6; ++ks) {
                Frag b;
#pragma unroll
                for (int v = 0; v < 8; ++v) {
                    const int kk = ks * 32 + ((v >> 2) << 4) + hl * 8 + ((v & 3) << 1);
                    b.u[v] = *(const u32*)&lK[(nt * 16 + ln) * 200 + kk];
                }
                s4[nt] = wmma_bf16(aq[ks], b, s4[nt]);
            }

        // ---- online softmax (fp32) ----
        const int rowbase = q0 + wid * 16 + hl * 8;
#pragma unroll
        for (int r = 0; r < 8; ++r) {
            const int row = rowbase + r;
            float mx = m_i[r];
#pragma unroll
            for (int nt = 0; nt < 4; ++nt) {
                float v = s4[nt][r] * scale;
                const int col = kb * 64 + nt * 16 + ln;
                if (col > row) v = -1e30f;      // causal mask
                s4[nt][r] = v;
                mx = fmaxf(mx, v);
            }
            mx = half_max(mx);
            const float fsc = __expf(m_i[r] - mx);
            m_i[r] = mx;
            float rs = 0.f;
#pragma unroll
            for (int nt = 0; nt < 4; ++nt) {
                const float p = __expf(s4[nt][r] - mx);
                s4[nt][r] = p;
                rs += p;
            }
            l_i[r] = l_i[r] * fsc + half_sum(rs);
#pragma unroll
            for (int nt = 0; nt < 4; ++nt)
                lP[(wid * 16 + hl * 8 + r) * 68 + nt * 16 + ln] = f2bf(s4[nt][r]);
#pragma unroll
            for (int a = 0; a < 8; ++a) o_acc[a][r] *= fsc;
        }

        // ---- O += P @ V ----
#pragma unroll
        for (int ks2 = 0; ks2 < 2; ++ks2) {
            Frag a;
#pragma unroll
            for (int v = 0; v < 8; ++v) {
                const int K0 = ks2 * 32 + ((v >> 2) << 4) + hl * 8 + ((v & 3) << 1);
                a.u[v] = *(const u32*)&lP[(wid * 16 + ln) * 68 + K0];
            }
#pragma unroll
            for (int dt = 0; dt < 8; ++dt) {
                Frag b;
#pragma unroll
                for (int v = 0; v < 8; ++v) {
                    const int kk = ks2 * 32 + ((v >> 2) << 4) + hl * 8 + ((v & 3) << 1);
                    b.u[v] = *(const u32*)&lVT[(dt * 16 + ln) * 68 + kk];
                }
                o_acc[dt] = wmma_bf16(a, b, o_acc[dt]);
            }
        }
    }

    // ---- epilogue: O / l -> bf16 ----
#pragma unroll
    for (int dt = 0; dt < 8; ++dt)
#pragma unroll
        for (int r = 0; r < 8; ++r) {
            const int row = q0 + wid * 16 + hl * 8 + r;
            const int col = h * 128 + dt * 16 + ln;
            O[(size_t)row * 2048 + col] = f2bf(o_acc[dt][r] / l_i[r]);
        }
}

// ---------------------------------------------------------------------------
// Host-side orchestration
// ---------------------------------------------------------------------------
extern "C" void kernel_launch(void* const* d_in, const int* in_sizes, int n_in,
                              void* d_out, int out_size, void* d_ws, size_t ws_size,
                              hipStream_t stream) {
    (void)in_sizes; (void)n_in; (void)out_size; (void)ws_size;
    const int S = 2048, H = 2048, QL = 1536, KVL = 512, NH = 16;
    const int QKH = 192, NDV = 256, QW = NH * QKH /*3072*/, KVW = NH * NDV /*4096*/;
    const int KVA = KVL + 64 /*576*/;

    const float* hidden = (const float*)d_in[0];
    // d_in[1] = attention_mask (causal; applied analytically)
    const float* Wqa  = (const float*)d_in[2];
    const float* gqa  = (const float*)d_in[3];
    const float* Wqb  = (const float*)d_in[4];
    const float* Wkva = (const float*)d_in[5];
    const float* gkva = (const float*)d_in[6];
    const float* Wkvb = (const float*)d_in[7];
    const float* Wo   = (const float*)d_in[8];
    float* out = (float*)d_out;

    char* w = (char*)d_ws;
    auto carve = [&](size_t bytes) -> void* {
        void* p = (void*)w;
        w += (bytes + 255) & ~(size_t)255;
        return p;
    };
    u16* hb     = (u16*)carve((size_t)S * H * 2);        // hidden bf16 (row-major)
    u16* wqa_t  = (u16*)carve((size_t)H * QL * 2);       // all weights bf16, TRANSPOSED
    u16* wqb_t  = (u16*)carve((size_t)QL * QW * 2);
    u16* wkva_t = (u16*)carve((size_t)H * KVA * 2);
    u16* wkvb_t = (u16*)carve((size_t)KVL * KVW * 2);
    u16* wo_t   = (u16*)carve((size_t)(NH * 128) * H * 2);
    float* qc   = (float*)carve((size_t)S * QL * 4);     // hidden@Wqa (fp32)
    u16* qcn    = (u16*)carve((size_t)S * QL * 2);       // rmsnorm(qc) bf16
    u16* qmat   = (u16*)carve((size_t)S * QW * 2);       // q per head
    float* kvc  = (float*)carve((size_t)S * KVA * 4);    // hidden@Wkva (fp32)
    u16* kvan   = (u16*)carve((size_t)S * KVL * 2);      // rmsnorm(kv_a) bf16
    u16* krope  = (u16*)carve((size_t)S * 64 * 2);
    u16* kvmat  = (u16*)carve((size_t)S * KVW * 2);      // [k_nope|v] per head
    u16* attn_b = (u16*)carve((size_t)S * (NH * 128) * 2);

    {   // activations: plain convert
        const int n4 = (S * H) / 4;
        cvt_f32_bf16_kernel<<<(n4 + 255) / 256, 256, 0, stream>>>(hidden, hb, n4);
    }
    auto cvtT = [&](const float* src, u16* dst, int K, int N) {
        cvt_transpose_kernel<<<dim3(N / 32, K / 32), 256, 0, stream>>>(src, dst, K, N);
    };
    cvtT(Wqa,  wqa_t,  H,   QL);
    cvtT(Wqb,  wqb_t,  QL,  QW);
    cvtT(Wkva, wkva_t, H,   KVA);
    cvtT(Wkvb, wkvb_t, KVL, KVW);
    cvtT(Wo,   wo_t,   NH * 128, H);

    // q_c = hidden @ Wqa ; kv_c = hidden @ Wkva
    gemm_bf16_kernel<false><<<dim3(QL / 64, S / 128), 256, 0, stream>>>(hb, wqa_t, qc, S, QL, H);
    gemm_bf16_kernel<false><<<dim3(KVA / 64, S / 128), 256, 0, stream>>>(hb, wkva_t, kvc, S, KVA, H);

    rmsnorm_kernel<<<S, 256, 0, stream>>>(qc, gqa, qcn, QL, QL);
    rmsnorm_kernel<<<S, 256, 0, stream>>>(kvc, gkva, kvan, KVL, KVA);
    krope_kernel<<<(S * 64) / 256, 256, 0, stream>>>(kvc, krope);

    // q = q_cn @ Wqb ; kv = kv_an @ Wkvb
    gemm_bf16_kernel<true><<<dim3(QW / 64, S / 128), 256, 0, stream>>>(qcn, wqb_t, qmat, S, QW, QL);
    gemm_bf16_kernel<true><<<dim3(KVW / 64, S / 128), 256, 0, stream>>>(kvan, wkvb_t, kvmat, S, KVW, KVL);

    // flash attention
    mla_attn_kernel<<<dim3(S / 128, NH), 256, 0, stream>>>(qmat, kvmat, krope, attn_b);

    // out = attn_out @ Wo
    gemm_bf16_kernel<false><<<dim3(H / 64, S / 128), 256, 0, stream>>>(attn_b, wo_t, out, S, H, NH * 128);
}
